// Critic_71004399337709
// MI455X (gfx1250) — matrix operand order repaired and checked by reading, not compiled, not run on hardware
//
#include <hip/hip_runtime.h>
#include <hip/hip_bf16.h>

// ---------------------------------------------------------------------------
// CDNA5 (gfx1250) GNN pipeline: edge MLP via v_wmma_f32_16x16x32_bf16,
// scatter via global_atomic_add_f32 (L2-resident agg), GRU gates via WMMA.
// ---------------------------------------------------------------------------

typedef __attribute__((ext_vector_type(16))) __bf16 v16bf;
typedef __attribute__((ext_vector_type(8)))  __bf16 v8bf;
typedef __attribute__((ext_vector_type(4)))  __bf16 v4bf;
typedef __attribute__((ext_vector_type(8)))  float  v8f;
typedef __attribute__((ext_vector_type(4)))  float  v4f;

union bfu { v16bf v; v8bf h[2]; __bf16 e[16]; };

__device__ __forceinline__ float selu_f(float x) {
    const float a = 1.6732632423543772f, l = 1.0507009873554805f;
    return l * (x > 0.f ? x : a * (__expf(x) - 1.f));
}
__device__ __forceinline__ float sigm_f(float x) { return 1.f / (1.f + __expf(-x)); }
__device__ __forceinline__ float tanh_f(float x) {
    float xc = fminf(15.f, fmaxf(-15.f, x));
    float e = __expf(2.f * xc);
    return (e - 1.f) / (e + 1.f);
}

__device__ __forceinline__ v16bf ldb(const __bf16* p) { return *(const v16bf*)p; }

__device__ __forceinline__ v8f wmma_bf16(const v16bf& a, const v16bf& b, const v8f& c) {
    return __builtin_amdgcn_wmma_f32_16x16x32_bf16(false, a, false, b, (short)0, c, false, false);
}

// ---------------------------------------------------------------------------
// Swizzle weights into WMMA B-operand layout (bf16).
// B tile (K=32 x N=16): swz[(tile*32+lane)*16 + j] = W[nt*16+(lane&15)][kc*32+(lane>>4)*8+(j&7)+(j>>3)*16]
// W_msg: [64][128] -> 16 tiles (nt 0..3, kc 0..3), tile = nt*4+kc
// W_ih/W_hh: [192][64] -> 24 tiles (nt 0..11, kc 0..1), tile = nt*2+kc
// ---------------------------------------------------------------------------
__global__ void prep_weights(const float* __restrict__ W_msg,
                             const float* __restrict__ W_ih,
                             const float* __restrict__ W_hh,
                             __bf16* __restrict__ wmsg_sw,
                             __bf16* __restrict__ wih_sw,
                             __bf16* __restrict__ whh_sw) {
    int idx = blockIdx.x * blockDim.x + threadIdx.x;  // one (tile,lane) per thread
    if (idx < 16 * 32) {
        int tile = idx >> 5, lane = idx & 31;
        int nt = tile >> 2, kc = tile & 3;
        int n = nt * 16 + (lane & 15);
        int kb = kc * 32 + (lane >> 4) * 8;
        #pragma unroll
        for (int j = 0; j < 16; ++j)
            wmsg_sw[(tile * 32 + lane) * 16 + j] = (__bf16)W_msg[n * 128 + kb + (j & 7) + (j >> 3) * 16];
    }
    if (idx < 24 * 32) {
        int tile = idx >> 5, lane = idx & 31;
        int nt = tile >> 1, kc = tile & 1;
        int n = nt * 16 + (lane & 15);
        int kb = kc * 32 + (lane >> 4) * 8;
        #pragma unroll
        for (int j = 0; j < 16; ++j) {
            wih_sw[(tile * 32 + lane) * 16 + j] = (__bf16)W_ih[n * 64 + kb + (j & 7) + (j >> 3) * 16];
            whh_sw[(tile * 32 + lane) * 16 + j] = (__bf16)W_hh[n * 64 + kb + (j & 7) + (j >> 3) * 16];
        }
    }
}

// ----- vectorized utility sweeps (n4 = element count / 4) ------------------
__global__ void copy_f32v(const float* __restrict__ src, float* __restrict__ dst, long n4) {
    long i = (long)blockIdx.x * blockDim.x + threadIdx.x;
    long s = (long)gridDim.x * blockDim.x;
    for (; i < n4; i += s) ((v4f*)dst)[i] = ((const v4f*)src)[i];
}
__global__ void zero_f32v(float* __restrict__ p, long n4) {
    long i = (long)blockIdx.x * blockDim.x + threadIdx.x;
    long s = (long)gridDim.x * blockDim.x;
    v4f z = {};
    for (; i < n4; i += s) ((v4f*)p)[i] = z;
}
__global__ void f32_to_bf16v(const float* __restrict__ src, __bf16* __restrict__ dst, long n4) {
    long i = (long)blockIdx.x * blockDim.x + threadIdx.x;
    long s = (long)gridDim.x * blockDim.x;
    for (; i < n4; i += s) {
        v4f x = ((const v4f*)src)[i];
        ((v4bf*)dst)[i] = __builtin_convertvector(x, v4bf);
    }
}
__global__ void zero_f32(float* __restrict__ p, long n) {
    long i = (long)blockIdx.x * blockDim.x + threadIdx.x;
    if (i < n) p[i] = 0.f;
}

// ---------------------------------------------------------------------------
// Edge message kernel. 1 wave = 16 edges x 64 outputs = 16 WMMAs.
// A tiles: per lane two b128 loads at feature offsets (c + sub*8), (c+16+sub*8).
// D (16x16 f32): element (M = r + 8*sub, N = lane&15). Scatter-add to agg.
// ---------------------------------------------------------------------------
__global__ void __launch_bounds__(256)
edge_msg(const __bf16* __restrict__ stateb, const int* __restrict__ first,
         const int* __restrict__ second, const __bf16* __restrict__ wmsg_sw,
         const float* __restrict__ b_msg, float* __restrict__ agg,
         int n_tiles, int n_edges) {
    int lane = threadIdx.x & 31;
    int tile = blockIdx.x * (blockDim.x >> 5) + (threadIdx.x >> 5);
    if (tile >= n_tiles) return;  // wave-uniform: EXEC all-ones at every WMMA
    int ebase = tile * 16;
    int m = lane & 15, sub = lane >> 4;

    // warm L1 with this wave's B-operand region (global_prefetch_b8)
    __builtin_prefetch(wmsg_sw + lane * 16, 0, 3);

    int ei = ebase + m; if (ei >= n_edges) ei = n_edges - 1;
    int rowF = first[ei];
    int rowS = second[ei];

    bfu a[4];
    const __bf16* pf = stateb + (long)rowF * 64 + sub * 8;
    const __bf16* ps = stateb + (long)rowS * 64 + sub * 8;
    a[0].h[0] = *(const v8bf*)(pf);       a[0].h[1] = *(const v8bf*)(pf + 16);
    a[1].h[0] = *(const v8bf*)(pf + 32);  a[1].h[1] = *(const v8bf*)(pf + 48);
    a[2].h[0] = *(const v8bf*)(ps);       a[2].h[1] = *(const v8bf*)(ps + 16);
    a[3].h[0] = *(const v8bf*)(ps + 32);  a[3].h[1] = *(const v8bf*)(ps + 48);

    // Pre-scaled element offset of 'second' row for each C-element row.
    int rsoff[8];
    int rowS64 = rowS << 6;
    #pragma unroll
    for (int r = 0; r < 8; ++r) rsoff[r] = __shfl(rowS64, r + 8 * sub, 32);

    // wave-uniform "all 16 edges valid" predicate in an SGPR
    int full = __builtin_amdgcn_readfirstlane((ebase + 16 <= n_edges) ? 1 : 0);

    #pragma unroll
    for (int nt = 0; nt < 4; ++nt) {
        v8f c = {};
        #pragma unroll
        for (int kc = 0; kc < 4; ++kc) {
            v16bf b = ldb(wmsg_sw + ((nt * 4 + kc) * 32 + lane) * 16);
            c = wmma_bf16(a[kc].v, b, c);
        }
        float bias = b_msg[nt * 16 + m];
        float* aggcol = agg + nt * 16 + m;
        if (full) {
            #pragma unroll
            for (int r = 0; r < 8; ++r) {
                float s = selu_f(c[r] + bias);
                __hip_atomic_fetch_add(aggcol + rsoff[r], s,
                                       __ATOMIC_RELAXED, __HIP_MEMORY_SCOPE_AGENT);
            }
        } else {
            #pragma unroll
            for (int r = 0; r < 8; ++r) {
                if (ebase + r + 8 * sub < n_edges) {
                    float s = selu_f(c[r] + bias);
                    __hip_atomic_fetch_add(aggcol + rsoff[r], s,
                                           __ATOMIC_RELAXED, __HIP_MEMORY_SCOPE_AGENT);
                }
            }
        }
    }
}

// ---------------------------------------------------------------------------
// GRU update. 1 wave = 16 nodes. x = agg (f32 -> bf16 on the fly), h from
// bf16 state shadow; 12 WMMAs per 16-column group produce all six gate tiles.
// ---------------------------------------------------------------------------
__global__ void __launch_bounds__(256)
gru_update(const float* __restrict__ agg, float* __restrict__ state,
           const __bf16* __restrict__ stateb,
           const __bf16* __restrict__ wih_sw, const __bf16* __restrict__ whh_sw,
           const float* __restrict__ b_ih, const float* __restrict__ b_hh,
           int n_tiles, int n_nodes) {
    int lane = threadIdx.x & 31;
    int tile = blockIdx.x * (blockDim.x >> 5) + (threadIdx.x >> 5);
    if (tile >= n_tiles) return;  // wave-uniform
    int nbase = tile * 16;
    int m = lane & 15, sub = lane >> 4;
    int rowi = nbase + m; if (rowi >= n_nodes) rowi = n_nodes - 1;
    long row = rowi;

    int safe = __builtin_amdgcn_readfirstlane((nbase + 16 <= n_nodes) ? 1 : 0);

    // A for x = agg (convert f32 -> bf16 into A-operand layout)
    bfu ax[2];
    const float* pax = agg + row * 64 + sub * 8;
    #pragma unroll
    for (int kc = 0; kc < 2; ++kc) {
        #pragma unroll
        for (int j = 0; j < 8; ++j) {
            ax[kc].e[j]     = (__bf16)pax[kc * 32 + j];
            ax[kc].e[j + 8] = (__bf16)pax[kc * 32 + 16 + j];
        }
    }
    // A for h (bf16 shadow, direct b128 loads)
    bfu ah[2];
    const __bf16* pah = stateb + row * 64 + sub * 8;
    ah[0].h[0] = *(const v8bf*)(pah);       ah[0].h[1] = *(const v8bf*)(pah + 16);
    ah[1].h[0] = *(const v8bf*)(pah + 32);  ah[1].h[1] = *(const v8bf*)(pah + 48);

    #pragma unroll
    for (int f = 0; f < 4; ++f) {
        v8f ir = {}, iz = {}, in_ = {}, hr = {}, hz = {}, hn = {};
        #pragma unroll
        for (int kc = 0; kc < 2; ++kc) {
            // gate g occupies ntiles g*4+f ; tileIdx = nt*2 + kc
            ir  = wmma_bf16(ax[kc].v, ldb(wih_sw + (((0 * 4 + f) * 2 + kc) * 32 + lane) * 16), ir);
            iz  = wmma_bf16(ax[kc].v, ldb(wih_sw + (((1 * 4 + f) * 2 + kc) * 32 + lane) * 16), iz);
            in_ = wmma_bf16(ax[kc].v, ldb(wih_sw + (((2 * 4 + f) * 2 + kc) * 32 + lane) * 16), in_);
            hr  = wmma_bf16(ah[kc].v, ldb(whh_sw + (((0 * 4 + f) * 2 + kc) * 32 + lane) * 16), hr);
            hz  = wmma_bf16(ah[kc].v, ldb(whh_sw + (((1 * 4 + f) * 2 + kc) * 32 + lane) * 16), hz);
            hn  = wmma_bf16(ah[kc].v, ldb(whh_sw + (((2 * 4 + f) * 2 + kc) * 32 + lane) * 16), hn);
        }
        int col = f * 16 + m;
        float bir = b_ih[col], biz = b_ih[64 + col], bin_ = b_ih[128 + col];
        float bhr = b_hh[col], bhz = b_hh[64 + col], bhn_ = b_hh[128 + col];
        float* sp = state + (long)(nbase + 8 * sub) * 64 + col;
        if (safe) {
            #pragma unroll
            for (int r = 0; r < 8; ++r) {
                float hv = sp[r * 64];
                float rr = sigm_f(ir[r] + bir + hr[r] + bhr);
                float zz = sigm_f(iz[r] + biz + hz[r] + bhz);
                float nn = tanh_f(in_[r] + bin_ + rr * (hn[r] + bhn_));
                sp[r * 64] = (1.f - zz) * nn + zz * hv;
            }
        } else {
            #pragma unroll
            for (int r = 0; r < 8; ++r) {
                int node = nbase + r + 8 * sub;
                if (node < n_nodes) {
                    float hv = sp[r * 64];
                    float rr = sigm_f(ir[r] + bir + hr[r] + bhr);
                    float zz = sigm_f(iz[r] + biz + hz[r] + bhz);
                    float nn = tanh_f(in_[r] + bin_ + rr * (hn[r] + bhn_));
                    sp[r * 64] = (1.f - zz) * nn + zz * hv;
                }
            }
        }
    }
}

// ---------------------------------------------------------------------------
// feature[f] = sum over nodes of state[:, f]
// ---------------------------------------------------------------------------
__global__ void __launch_bounds__(256)
reduce_feature(const float* __restrict__ state, float* __restrict__ feature, int n_nodes) {
    int t = threadIdx.x;
    int feat = t & 63;
    int grp = t >> 6;  // 0..3
    float acc = 0.f;
    for (long node = (long)blockIdx.x * 4 + grp; node < n_nodes; node += (long)gridDim.x * 4)
        acc += state[node * 64 + feat];
    __hip_atomic_fetch_add(feature + feat, acc, __ATOMIC_RELAXED, __HIP_MEMORY_SCOPE_AGENT);
}

// ---------------------------------------------------------------------------
// Readout MLP: selu(W_r1 f + b1) -> selu(W_r2 h + b2) -> W_out h + b_out
// ---------------------------------------------------------------------------
__global__ void __launch_bounds__(64)
readout(const float* __restrict__ feature,
        const float* __restrict__ W_r1, const float* __restrict__ b_r1,
        const float* __restrict__ W_r2, const float* __restrict__ b_r2,
        const float* __restrict__ W_out, const float* __restrict__ b_out,
        float* __restrict__ out) {
    __shared__ float f0[64];
    __shared__ float h1[64];
    __shared__ float h2[64];
    int t = threadIdx.x;
    f0[t] = feature[t];
    __syncthreads();
    float acc = b_r1[t];
    for (int k = 0; k < 64; ++k) acc += W_r1[t * 64 + k] * f0[k];
    h1[t] = selu_f(acc);
    __syncthreads();
    float acc2 = b_r2[t];
    for (int k = 0; k < 64; ++k) acc2 += W_r2[t * 64 + k] * h1[k];
    h2[t] = selu_f(acc2);
    __syncthreads();
    if (t == 0) {
        float o = b_out[0];
        for (int k = 0; k < 64; ++k) o += W_out[k] * h2[k];
        out[0] = o;
    }
}

// ---------------------------------------------------------------------------
extern "C" void kernel_launch(void* const* d_in, const int* in_sizes, int n_in,
                              void* d_out, int out_size, void* d_ws, size_t ws_size,
                              hipStream_t stream) {
    (void)n_in; (void)out_size; (void)ws_size;
    const float* link_state = (const float*)d_in[0];
    const int*   first      = (const int*)d_in[1];
    const int*   second     = (const int*)d_in[2];
    const float* W_msg      = (const float*)d_in[3];
    const float* b_msg      = (const float*)d_in[4];
    const float* W_ih       = (const float*)d_in[5];
    const float* b_ih       = (const float*)d_in[6];
    const float* W_hh       = (const float*)d_in[7];
    const float* b_hh       = (const float*)d_in[8];
    const float* W_r1       = (const float*)d_in[9];
    const float* b_r1       = (const float*)d_in[10];
    const float* W_r2       = (const float*)d_in[11];
    const float* b_r2       = (const float*)d_in[12];
    const float* W_out_     = (const float*)d_in[13];
    const float* b_out_     = (const float*)d_in[14];

    const int NL = in_sizes[0] / 64;   // 100000
    const int NE = in_sizes[1];        // 2000000
    const long nel = (long)NL * 64;

    // workspace carve-up (all 256B-aligned sections)
    char* ws = (char*)d_ws;
    float*  state   = (float*)ws;   ws += ((nel * 4 + 255) & ~255L);
    float*  agg     = (float*)ws;   ws += ((nel * 4 + 255) & ~255L);
    __bf16* stateb  = (__bf16*)ws;  ws += ((nel * 2 + 255) & ~255L);
    __bf16* wmsg_sw = (__bf16*)ws;  ws += ((16 * 32 * 16 * 2 + 255) & ~255L);
    __bf16* wih_sw  = (__bf16*)ws;  ws += ((24 * 32 * 16 * 2 + 255) & ~255L);
    __bf16* whh_sw  = (__bf16*)ws;  ws += ((24 * 32 * 16 * 2 + 255) & ~255L);
    float*  feature = (float*)ws;   ws += 256;

    const int etiles = (NE + 15) / 16;
    const int eblocks = (etiles + 7) / 8;
    const int ntiles = (NL + 15) / 16;
    const int nblocks = (ntiles + 7) / 8;
    const long n4 = nel / 4;

    prep_weights<<<3, 256, 0, stream>>>(W_msg, W_ih, W_hh, wmsg_sw, wih_sw, whh_sw);
    copy_f32v<<<1024, 256, 0, stream>>>(link_state, state, n4);

    for (int t = 0; t < 4; ++t) {
        f32_to_bf16v<<<1024, 256, 0, stream>>>(state, stateb, n4);
        zero_f32v<<<1024, 256, 0, stream>>>(agg, n4);
        edge_msg<<<eblocks, 256, 0, stream>>>(stateb, first, second, wmsg_sw, b_msg,
                                              agg, etiles, NE);
        gru_update<<<nblocks, 256, 0, stream>>>(agg, state, stateb, wih_sw, whh_sw,
                                                b_ih, b_hh, ntiles, NL);
    }

    zero_f32<<<1, 64, 0, stream>>>(feature, 64);
    reduce_feature<<<256, 256, 0, stream>>>(state, feature, NL);
    readout<<<1, 64, 0, stream>>>(feature, W_r1, b_r1, W_r2, b_r2, W_out_, b_out_,
                                  (float*)d_out);
}
